// GATLayer_8040178778933
// MI455X (gfx1250) — matrix operand order repaired and checked by reading, not compiled, not run on hardware
//
#include <hip/hip_runtime.h>

typedef _Float16 v16h  __attribute__((ext_vector_type(16)));
typedef float    v8f   __attribute__((ext_vector_type(8)));
typedef _Float16 half8 __attribute__((ext_vector_type(8)));
typedef float    f4    __attribute__((ext_vector_type(4)));
typedef int      i4    __attribute__((ext_vector_type(4)));

#define GN       4096
#define GF_IN    256
#define GHEADS   8
#define GFEAT    64
#define GHOUT    512   // GHEADS*GFEAT
#define GNEG     0.2f

// gfx1250 async global->LDS copies (ASYNCcnt path). Guarded: falls back to
// synchronous VGPR-staged copies if the toolchain lacks the builtins.
// Probe-derived signatures:
//   b128: (int4* gsrc, addrspace(3) int4* ldst, imm offset, imm cpol)
//   b32 : (int*  gsrc, addrspace(3) int*  ldst, imm offset, imm cpol)
#if __has_builtin(__builtin_amdgcn_global_load_async_to_lds_b128) && \
    __has_builtin(__builtin_amdgcn_global_load_async_to_lds_b32)  && \
    __has_builtin(__builtin_amdgcn_s_wait_asynccnt)
#define USE_ASYNC_LDS 1
typedef __attribute__((address_space(3))) i4  lds_i4;
typedef __attribute__((address_space(3))) int lds_int;
#else
#define USE_ASYNC_LDS 0
#endif

// ---------------------------------------------------------------------------
// Kernel A: h_t = (x @ W)^T stored feature-major: h_t[c][n], c = head*64+f.
// WMMA C/D layout hands each lane 8 consecutive M-rows of one N-column, so
// the transposed store is a single b128 per accumulator tile column.
// ---------------------------------------------------------------------------
__global__ __launch_bounds__(128) void gat_gemm_h(
    const float* __restrict__ x, const float* __restrict__ W,
    _Float16* __restrict__ h_t) {
  __shared__ _Float16 xs[64 * 34];   // [m][k], padded stride 34
  __shared__ _Float16 wt[64 * 34];   // [n][k] (W tile transposed), padded

  const int t    = threadIdx.x;
  const int lane = t & 31;
  const int wv   = t >> 5;           // wave id 0..3 -> M sub-tile
  const int l15  = lane & 15;
  const int kA   = (lane >> 4) * 8;  // A-operand K base (ISA 16-bit A layout)
  const int kB   = (lane >> 4) * 16; // unused; B operand is read with kA, same k order as A
  const int i0   = blockIdx.x * 64;
  const int n0   = blockIdx.y * 64;

  v8f acc[4];
#pragma unroll
  for (int nt = 0; nt < 4; ++nt)
#pragma unroll
    for (int e = 0; e < 8; ++e) acc[nt][e] = 0.0f;

  for (int k0 = 0; k0 < GF_IN; k0 += 32) {
    __syncthreads();
#pragma unroll
    for (int r = 0; r < 16; ++r) {          // x tile: 64x32 f32 -> f16
      int lin = r * 128 + t;
      int row = lin >> 5, col = lin & 31;
      xs[row * 34 + col] = (_Float16)x[(size_t)(i0 + row) * GF_IN + k0 + col];
    }
#pragma unroll
    for (int r = 0; r < 16; ++r) {          // W tile: 32x64, store transposed
      int lin = r * 128 + t;
      int kk = lin >> 6, col = lin & 63;
      wt[col * 34 + kk] = (_Float16)W[(size_t)(k0 + kk) * GHOUT + n0 + col];
    }
    __syncthreads();

    v16h Aop;
    const int am = wv * 16 + l15;
#pragma unroll
    for (int v = 0; v < 8; ++v) {
      int kk = (v >> 2) * 16 + kA + (v & 3) * 2;
      Aop[2 * v]     = xs[am * 34 + kk];
      Aop[2 * v + 1] = xs[am * 34 + kk + 1];
    }
#pragma unroll
    for (int nt = 0; nt < 4; ++nt) {
      const int bn = nt * 16 + l15;
      v16h Bop;
#pragma unroll
      for (int v = 0; v < 8; ++v) {
        int kk = (v >> 2) * 16 + kA + (v & 3) * 2;
        Bop[2 * v]     = wt[bn * 34 + kk];
        Bop[2 * v + 1] = wt[bn * 34 + kk + 1];
      }
      acc[nt] = __builtin_amdgcn_wmma_f32_16x16x32_f16(
          false, Aop, false, Bop, (short)0, acc[nt], false, false);
    }
  }

  // Transposed store: lane holds rows (mhi..mhi+7) of column col -> one b128.
  const int mhi  = (lane & 16) ? 8 : 0;
  const int row0 = i0 + wv * 16 + mhi;
#pragma unroll
  for (int nt = 0; nt < 4; ++nt) {
    const int col = n0 + nt * 16 + l15;
    half8 hv;
#pragma unroll
    for (int r = 0; r < 8; ++r) hv[r] = (_Float16)acc[nt][r];
    *(half8*)(h_t + (size_t)col * GN + row0) = hv;
  }
}

// ---------------------------------------------------------------------------
// Kernel B: s_src[n,h] = h[n,h,:]·a_src ; s_dst likewise. Coalesced h_t rows.
// ---------------------------------------------------------------------------
__global__ __launch_bounds__(256) void gat_sdots(
    const _Float16* __restrict__ h_t, const float* __restrict__ a_src,
    const float* __restrict__ a_dst, float* __restrict__ ssrc,
    float* __restrict__ sdst) {
  const int t  = blockIdx.x * 256 + threadIdx.x;  // 0..N*HEADS-1
  const int hh = t >> 12;                         // head (constant per block)
  const int n  = t & (GN - 1);                    // node (coalesced)
  float ss = 0.f, sd = 0.f;
#pragma unroll 8
  for (int f = 0; f < GFEAT; ++f) {
    const float hv = (float)h_t[(size_t)(hh * GFEAT + f) * GN + n];
    ss += hv * a_src[f];
    sd += hv * a_dst[f];
  }
  ssrc[n * GHEADS + hh] = ss;
  sdst[n * GHEADS + hh] = sd;
}

// ---------------------------------------------------------------------------
// Kernel C: flash-style masked-softmax aggregation.
// Block = 32 output rows x all 8 heads; 8 waves, wave w = head w.
// j streamed in tiles of 32; tile fills go global->LDS via the async path
// (ASYNCcnt) when available; alpha built directly into WMMA A-operands with
// the denominator accumulated in the same pass.
// ---------------------------------------------------------------------------
__global__ __launch_bounds__(256) void gat_attn_agg(
    const float* __restrict__ adj, const _Float16* __restrict__ h_t,
    const float* __restrict__ ssrc_g, const float* __restrict__ sdst_g,
    float* __restrict__ out) {
  __shared__ float    adj_s[32 * 36];        // [i][j], stride 36 (16B aligned)
  __shared__ _Float16 ht_s[GHOUT * 40];      // [c][j], stride 40 (16B aligned)
  __shared__ float    sdst_s[32 * GHEADS];
  __shared__ float    ssrc_s[32 * GHEADS];

  const int t    = threadIdx.x;
  const int lane = t & 31;
  const int hh   = t >> 5;          // head for this wave
  const int l15  = lane & 15;
  const int kA   = (lane >> 4) * 8;
  const int kB   = (lane >> 4) * 16;
  const int i0   = blockIdx.x * 32;

  const int arow = t >> 3;          // adj tile row   (0..31)
  const int aq   = t & 7;           // adj float4 col (0..7)

  ssrc_s[t] = ssrc_g[(size_t)i0 * GHEADS + t];
  __syncthreads();
  float ss_reg[2];
#pragma unroll
  for (int mt = 0; mt < 2; ++mt)
    ss_reg[mt] = ssrc_s[(mt * 16 + l15) * GHEADS + hh];

  v8f acc[2][4];
#pragma unroll
  for (int mt = 0; mt < 2; ++mt)
#pragma unroll
    for (int nt = 0; nt < 4; ++nt)
#pragma unroll
      for (int e = 0; e < 8; ++e) acc[mt][nt][e] = 0.0f;
  float denom[2] = {0.f, 0.f};

  for (int jt = 0; jt < GN / 32; ++jt) {
    const int j0 = jt * 32;
    __syncthreads();

#if USE_ASYNC_LDS
    // adj tile 32x32: one async b128 per thread, straight into LDS
    __builtin_amdgcn_global_load_async_to_lds_b128(
        (i4*)(adj + (size_t)(i0 + arow) * GN + j0 + aq * 4),
        (lds_i4*)(adj_s + arow * 36 + aq * 4), 0, 0);
    // s_dst tile 32x8
    __builtin_amdgcn_global_load_async_to_lds_b32(
        (int*)(sdst_g + (size_t)j0 * GHEADS + t),
        (lds_int*)(sdst_s + t), 0, 0);
    // h_t tile: 512 feature-rows x 32 nodes
#pragma unroll
    for (int r = 0; r < 8; ++r) {
      int lin = r * 256 + t;        // 0..2047
      int c = lin >> 2;             // 0..511
      int q = lin & 3;              // half8 within the 32-node window
      __builtin_amdgcn_global_load_async_to_lds_b128(
          (i4*)(h_t + (size_t)c * GN + j0 + q * 8),
          (lds_i4*)(ht_s + c * 40 + q * 8), 0, 0);
    }
#else
    *(f4*)(adj_s + arow * 36 + aq * 4) =
        *(const f4*)(adj + (size_t)(i0 + arow) * GN + j0 + aq * 4);
    sdst_s[t] = sdst_g[(size_t)j0 * GHEADS + t];
#pragma unroll
    for (int r = 0; r < 8; ++r) {
      int lin = r * 256 + t;
      int c = lin >> 2;
      int q = lin & 3;
      *(half8*)(ht_s + c * 40 + q * 8) =
          *(const half8*)(h_t + (size_t)c * GN + j0 + q * 8);
    }
#endif
    // Prefetch next tile (speculative; dropped if OOB)
    if (jt + 1 < GN / 32) {
      __builtin_prefetch(adj + (size_t)(i0 + arow) * GN + j0 + 32 + aq * 4, 0, 1);
      __builtin_prefetch(h_t + (size_t)(t * 2 + 0) * GN + j0 + 32, 0, 1);
      __builtin_prefetch(h_t + (size_t)(t * 2 + 1) * GN + j0 + 32, 0, 1);
    }
#if USE_ASYNC_LDS
    __builtin_amdgcn_s_wait_asynccnt(0);
#endif
    __syncthreads();

    // Hoist s_dst values for this lane's 16 j-slots (mt-independent)
    float sd_reg[16];
#pragma unroll
    for (int v = 0; v < 8; ++v) {
      int kk = (v >> 2) * 16 + kA + (v & 3) * 2;
      sd_reg[2 * v]     = sdst_s[kk * GHEADS + hh];
      sd_reg[2 * v + 1] = sdst_s[(kk + 1) * GHEADS + hh];
    }

    // Build alpha A-operands + denominators (ISA 16-bit A layout)
    v16h Aop[2];
#pragma unroll
    for (int mt = 0; mt < 2; ++mt) {
      const int il = mt * 16 + l15;
      const float ss = ss_reg[mt];
#pragma unroll
      for (int v = 0; v < 8; ++v) {
        int kk = (v >> 2) * 16 + kA + (v & 3) * 2;
#pragma unroll
        for (int p = 0; p < 2; ++p) {
          int jl = kk + p;
          float av = adj_s[il * 36 + jl];
          float e  = ss + sd_reg[2 * v + p];
          e = (e > 0.f) ? e : GNEG * e;                       // leaky_relu
          float w = (av != 0.f || (i0 + il) == (j0 + jl))     // adj + I mask
                        ? __expf(e) : 0.f;
          denom[mt] += w;
          Aop[mt][2 * v + p] = (_Float16)w;
        }
      }
    }
    // B-operand = h_t tile (ISA 16-bit B layout) and 8 WMMAs
#pragma unroll
    for (int nt = 0; nt < 4; ++nt) {
      const int f = nt * 16 + l15;
      v16h Bop;
#pragma unroll
      for (int v = 0; v < 8; ++v) {
        int kk = (v >> 2) * 16 + kA + (v & 3) * 2;
        Bop[2 * v]     = ht_s[(hh * GFEAT + f) * 40 + kk];
        Bop[2 * v + 1] = ht_s[(hh * GFEAT + f) * 40 + kk + 1];
      }
#pragma unroll
      for (int mt = 0; mt < 2; ++mt)
        acc[mt][nt] = __builtin_amdgcn_wmma_f32_16x16x32_f16(
            false, Aop[mt], false, Bop, (short)0, acc[mt][nt], false, false);
    }
  }

  // Lane pair (l, l+16) covers complementary j-halves of row l&15.
#pragma unroll
  for (int mt = 0; mt < 2; ++mt)
    denom[mt] += __shfl_xor(denom[mt], 16, 32);

  // Normalize + store. C/D VGPR r holds row (r | mhi); its denominator lives
  // in lane (r | mhi) -> broadcast via shfl, divide via v_rcp_f32.
  const int mhi = (lane & 16) ? 8 : 0;
#pragma unroll
  for (int mt = 0; mt < 2; ++mt) {
#pragma unroll
    for (int nt = 0; nt < 4; ++nt) {
      const int col = hh * GFEAT + nt * 16 + l15;
#pragma unroll
      for (int r = 0; r < 8; ++r) {
        float d = __shfl(denom[mt], mhi + r, 32);
        const int row = i0 + mt * 16 + mhi + r;
        out[(size_t)row * GHOUT + col] = acc[mt][nt][r] * __builtin_amdgcn_rcpf(d);
      }
    }
  }
}

// ---------------------------------------------------------------------------
extern "C" void kernel_launch(void* const* d_in, const int* in_sizes, int n_in,
                              void* d_out, int out_size, void* d_ws, size_t ws_size,
                              hipStream_t stream) {
  (void)in_sizes; (void)n_in; (void)out_size; (void)ws_size;
  const float* x     = (const float*)d_in[0];
  const float* adj   = (const float*)d_in[1];
  const float* W     = (const float*)d_in[2];
  const float* a_src = (const float*)d_in[3];
  const float* a_dst = (const float*)d_in[4];
  float* out = (float*)d_out;

  // Workspace layout: h_t (feature-major, 4 MB) | s_src (128 KB) | s_dst (128 KB)
  _Float16* h_t = (_Float16*)d_ws;
  float* ssrc = (float*)((char*)d_ws + (size_t)GHOUT * GN * sizeof(_Float16));
  float* sdst = ssrc + (size_t)GN * GHEADS;

  gat_gemm_h<<<dim3(GN / 64, GHOUT / 64), 128, 0, stream>>>(x, W, h_t);
  gat_sdots<<<(GN * GHEADS) / 256, 256, 0, stream>>>(h_t, a_src, a_dst, ssrc, sdst);
  gat_attn_agg<<<GN / 32, 256, 0, stream>>>(adj, h_t, ssrc, sdst, out);
}
